// DenoisingBlock_21741124452580
// MI455X (gfx1250) — matrix operand order repaired and checked by reading, not compiled, not run on hardware
//
#include <hip/hip_runtime.h>
#include <cstdint>
#include <cstddef>

// ---------------- problem dims ----------------
static constexpr int cL   = 512;
static constexpr int cND  = 384;
static constexpr int cPD  = 128;
static constexpr int cH   = 8;
static constexpr int cHD  = 48;
static constexpr int cNQK = 4;
static constexpr int cNV  = 8;
static constexpr int cOC  = cND + cH*cNV*3 + cH*cPD; // 1600
static constexpr size_t cLL = (size_t)cL*cL;         // 262144
static constexpr unsigned uLL = 262144u;

// ---------------- d_in index map (jax alphabetical tree flatten) ----------------
enum {
  IN_NODE=0, IN_PAIR=1,
  P_FRAME_NORM_B=2, P_FRAME_NORM_G=3, P_FRAME_UPD_B=4, P_FRAME_UPD_W=5,
  IPA_BK_PTS=6, IPA_BO=7, IPA_BQ_PTS=8, IPA_BV_PTS=9, IPA_HEAD_W=10,
  IPA_LN_B=11, IPA_LN_G=12, IPA_W_PAIR_BIAS=13, IPA_WK=14, IPA_WK_PTS=15,
  IPA_WO=16, IPA_WQ=17, IPA_WQ_PTS=18, IPA_WV=19, IPA_WV_PTS=20,
  P_IPA_NORM_B=21, P_IPA_NORM_G=22,
  PT_B=23, PT_B1=24, PT_B2=25, PT_G=26, PT_W1=27, PT_W2=28,
  TIME_B1=29, TIME_B2=30, TIME_W1=31, TIME_W2=32,
  NT_B=33, NT_B1=34, NT_B2=35, NT_G=36, NT_W1=37, NT_W2=38,
  TI_BL=39, TI_BLG=40, TI_BO=41, TI_BOG=42, TI_BR=43, TI_BRG=44,
  TI_FN_B=45, TI_FN_G=46, TI_LN_B=47, TI_LN_G=48,
  TI_WL=49, TI_WLG=50, TI_WO=51, TI_WOG=52, TI_WR=53, TI_WRG=54,
  TO_BL=55, TO_BLG=56, TO_BO=57, TO_BOG=58, TO_BR=59, TO_BRG=60,
  TO_FN_B=61, TO_FN_G=62, TO_LN_B=63, TO_LN_G=64,
  TO_WL=65, TO_WLG=66, TO_WO=67, TO_WOG=68, TO_WR=69, TO_WRG=70,
  IN_ROTS=71, IN_TEMBED=72, IN_TRANS=73
};

// ---------------- async global->LDS (CDNA5), guarded ----------------
#if defined(__has_builtin)
#  if __has_builtin(__builtin_amdgcn_global_load_async_to_lds_b64)
#    define ASYNC_LDS 1
#  endif
#endif
#ifndef ASYNC_LDS
#  define ASYNC_LDS 0
#endif

#if ASYNC_LDS
typedef int v2i_vs __attribute__((vector_size(8)));
__device__ inline void cp8_async(unsigned short* dst_lds, const unsigned short* src_glb){
  __builtin_amdgcn_global_load_async_to_lds_b64(
      (__attribute__((address_space(1))) v2i_vs*)src_glb,
      (__attribute__((address_space(3))) v2i_vs*)dst_lds, 0, 0);
}
__device__ inline void async_wait0(){
#if __has_builtin(__builtin_amdgcn_s_wait_asynccnt)
  __builtin_amdgcn_s_wait_asynccnt(0);
#else
  asm volatile("s_wait_asynccnt 0x0" ::: "memory");
#endif
}
#endif

// ---------------- WMMA plumbing ----------------
typedef __attribute__((ext_vector_type(16))) __bf16 v16bf;
typedef __attribute__((ext_vector_type(8)))  float  v8f;

union FragBF { v16bf v; unsigned short u[16]; };
union FragF  { v8f v; float f[8]; };

__device__ inline unsigned short f2bf(float f){
  unsigned int u = __float_as_uint(f);
  u += 0x7FFFu + ((u >> 16) & 1u);          // round-to-nearest-even
  return (unsigned short)(u >> 16);
}
__device__ inline float sigm(float x){ return 1.f/(1.f+__expf(-x)); }

__device__ inline v8f wmma_bf16(v16bf a, v16bf b, v8f c){
  return __builtin_amdgcn_wmma_f32_16x16x32_bf16(false, a, false, b, (short)0, c, false, false);
}

// A-frag (16x32 bf16) per ISA 7.12.2. LDS layout As[row][k], ld=32. -> 2x ds_load_b128.
__device__ inline v16bf load_a_frag(const unsigned short* As, int row0){
  int lane = threadIdx.x & 31;
  int row  = row0 + (lane & 15);
  int half = lane >> 4;
  const unsigned short* p = As + row*32 + (half << 3);
  FragBF fr;
#pragma unroll
  for (int e = 0; e < 8; ++e) fr.u[e] = p[e];
#pragma unroll
  for (int e = 0; e < 8; ++e) fr.u[8+e] = p[16+e];
  return fr.v;
}
// B-frag (32x16 bf16). LDS tile TRANSPOSED: Bsw[n][k], ld=32. -> 2x ds_load_b128.
__device__ inline v16bf load_b_frag(const unsigned short* Bsw, int col0){
  int lane = threadIdx.x & 31;
  int col  = col0 + (lane & 15);
  int half = lane >> 4;
  const unsigned short* p = Bsw + col*32 + (half << 3);
  FragBF fr;
#pragma unroll
  for (int e = 0; e < 8; ++e) fr.u[e] = p[e];
#pragma unroll
  for (int e = 0; e < 8; ++e) fr.u[8+e] = p[16+e];
  return fr.v;
}

// ---------------- generic bf16 GEMM: C(MxN f32) = act(A(MxK) @ B(KxN) + bias) ----------------
// Requirements (all call sites satisfy): M % 64 == 0, K % 32 == 0. N arbitrary (N%64!=0 -> scalar B path).
// OUTMODE: 0 = store f32, 1 = accumulate-add into f32, 2 = store bf16.  ACT: 0 none, 1 relu, 2 sigmoid.
template<bool BIAS, int ACT, int OUTMODE>
__global__ __launch_bounds__(128) void gemm_kernel(
    const unsigned short* __restrict__ A, const unsigned short* __restrict__ B,
    const float* __restrict__ bias, float* Cf, unsigned short* Cb,
    int M, int N, int K)
{
  __shared__ alignas(16) unsigned short As[64*32];   // [row][k]
  __shared__ alignas(16) unsigned short Bsw[64*32];  // [n][k]  (transposed tile)
  int tiles_n = (N + 63) >> 6;
  int tm = blockIdx.x / tiles_n, tn = blockIdx.x % tiles_n;
  int m0 = tm*64, n0 = tn*64;
  int wave = threadIdx.x >> 5;
  int wm = (wave >> 1)*32, wn = (wave & 1)*32;
  unsigned uK = (unsigned)K, uN = (unsigned)N;
  bool nvec = ((N & 63) == 0);
  v8f acc[4] = {};
  for (int k0 = 0; k0 < K; k0 += 32){
    // A tile: rows contiguous in k (no guards: M,K tile-multiples), 8B chunks
#pragma unroll
    for (int it = 0; it < 4; ++it){
      unsigned idx = (unsigned)threadIdx.x + 128u*(unsigned)it;
      unsigned r = idx >> 3, kq = (idx & 7) << 2;
      const unsigned short* src = &A[(unsigned)(m0 + (int)r)*uK + (unsigned)k0 + kq];
#if ASYNC_LDS
      cp8_async(&As[r*32 + kq], src);
#else
      *reinterpret_cast<uint2*>(&As[r*32 + kq]) = *reinterpret_cast<const uint2*>(src);
#endif
    }
    // B tile transposed into LDS (register transpose -> synchronous)
    if (nvec){
#pragma unroll
      for (int it = 0; it < 4; ++it){
        unsigned idx = (unsigned)threadIdx.x + 128u*(unsigned)it;
        unsigned kk = idx >> 4, nq = (idx & 15) << 2;
        uint2 w = *reinterpret_cast<const uint2*>(&B[(unsigned)(k0 + (int)kk)*uN + (unsigned)n0 + nq]);
        const unsigned short* ws = reinterpret_cast<const unsigned short*>(&w);
        Bsw[(nq+0)*32 + kk] = ws[0];
        Bsw[(nq+1)*32 + kk] = ws[1];
        Bsw[(nq+2)*32 + kk] = ws[2];
        Bsw[(nq+3)*32 + kk] = ws[3];
      }
    } else {
#pragma unroll
      for (int it = 0; it < 16; ++it){
        unsigned idx = (unsigned)threadIdx.x + 128u*(unsigned)it;
        unsigned kk = idx >> 6, nn = idx & 63;
        int gn = n0 + (int)nn;
        Bsw[nn*32 + kk] = (gn < N) ? B[(unsigned)(k0 + (int)kk)*uN + (unsigned)gn] : (unsigned short)0;
      }
    }
    if (k0 + 32 < K){
      __builtin_prefetch(A + (unsigned)(m0 + (int)(threadIdx.x & 63))*uK + (unsigned)(k0 + 32), 0, 1);
      __builtin_prefetch(B + (unsigned)(k0 + 32 + (int)(threadIdx.x >> 2))*uN + (unsigned)n0, 0, 1);
    }
#if ASYNC_LDS
    async_wait0();
#endif
    __syncthreads();
    v16bf a0 = load_a_frag(As, wm + 0);
    v16bf a1 = load_a_frag(As, wm + 16);
    v16bf b0 = load_b_frag(Bsw, wn + 0);
    v16bf b1 = load_b_frag(Bsw, wn + 16);
    acc[0] = wmma_bf16(a0, b0, acc[0]);
    acc[1] = wmma_bf16(a0, b1, acc[1]);
    acc[2] = wmma_bf16(a1, b0, acc[2]);
    acc[3] = wmma_bf16(a1, b1, acc[3]);
    __syncthreads();
  }
  int lane = threadIdx.x & 31, half = lane >> 4, coll = lane & 15;
#pragma unroll
  for (int t = 0; t < 4; ++t){
    int mbase = m0 + wm + (t >> 1)*16;
    int col   = n0 + wn + (t & 1)*16 + coll;
    if (col >= N) continue;
    FragF fr; fr.v = acc[t];
    float bv = BIAS ? bias[col] : 0.f;
#pragma unroll
    for (int r = 0; r < 8; ++r){
      int row = mbase + half*8 + r;
      float y = fr.f[r] + bv;
      if (ACT == 1) y = fmaxf(y, 0.f);
      else if (ACT == 2) y = sigm(y);
      unsigned offc = (unsigned)row*uN + (unsigned)col;
      if (OUTMODE == 0) Cf[offc] = y;
      else if (OUTMODE == 1) Cf[offc] += y;
      else Cb[offc] = f2bf(y);
    }
  }
}

// ---------------- gated projection (K=N=128, M=262144) ----------------
// G_t is stored CHANNEL-MAJOR: G_t[c*cLL + row]  (row = i*cL + k of the pair grid).
__global__ __launch_bounds__(128) void gemm_gated_kernel(
    const unsigned short* __restrict__ A, const unsigned short* __restrict__ B1,
    const unsigned short* __restrict__ B2, const float* __restrict__ b1,
    const float* __restrict__ b2, unsigned short* __restrict__ Gt, int M)
{
  __shared__ alignas(16) unsigned short As[64*32];
  __shared__ alignas(16) unsigned short B1sw[64*32];
  __shared__ alignas(16) unsigned short B2sw[64*32];
  int tm = blockIdx.x >> 1, tn = blockIdx.x & 1;
  int m0 = tm*64, n0 = tn*64;
  int wave = threadIdx.x >> 5;
  int wm = (wave >> 1)*32, wn = (wave & 1)*32;
  v8f acc1[4] = {}, acc2[4] = {};
  for (int k0 = 0; k0 < cPD; k0 += 32){
#pragma unroll
    for (int it = 0; it < 4; ++it){
      unsigned idx = (unsigned)threadIdx.x + 128u*(unsigned)it;
      unsigned r = idx >> 3, kq = (idx & 7) << 2;
      const unsigned short* src = &A[(unsigned)(m0 + (int)r)*(unsigned)cPD + (unsigned)k0 + kq];
#if ASYNC_LDS
      cp8_async(&As[r*32 + kq], src);
#else
      *reinterpret_cast<uint2*>(&As[r*32 + kq]) = *reinterpret_cast<const uint2*>(src);
#endif
    }
#pragma unroll
    for (int it = 0; it < 4; ++it){
      unsigned idx = (unsigned)threadIdx.x + 128u*(unsigned)it;
      unsigned kk = idx >> 4, nq = (idx & 15) << 2;
      unsigned goff = (unsigned)(k0 + (int)kk)*(unsigned)cPD + (unsigned)n0 + nq;
      uint2 w1 = *reinterpret_cast<const uint2*>(&B1[goff]);
      uint2 w2 = *reinterpret_cast<const uint2*>(&B2[goff]);
      const unsigned short* s1 = reinterpret_cast<const unsigned short*>(&w1);
      const unsigned short* s2 = reinterpret_cast<const unsigned short*>(&w2);
      B1sw[(nq+0)*32 + kk] = s1[0]; B1sw[(nq+1)*32 + kk] = s1[1];
      B1sw[(nq+2)*32 + kk] = s1[2]; B1sw[(nq+3)*32 + kk] = s1[3];
      B2sw[(nq+0)*32 + kk] = s2[0]; B2sw[(nq+1)*32 + kk] = s2[1];
      B2sw[(nq+2)*32 + kk] = s2[2]; B2sw[(nq+3)*32 + kk] = s2[3];
    }
    if (k0 + 32 < cPD)
      __builtin_prefetch(A + (unsigned)(m0 + (int)(threadIdx.x & 63))*(unsigned)cPD + (unsigned)(k0 + 32), 0, 1);
#if ASYNC_LDS
    async_wait0();
#endif
    __syncthreads();
    v16bf a0 = load_a_frag(As, wm + 0);
    v16bf a1 = load_a_frag(As, wm + 16);
    v16bf p0 = load_b_frag(B1sw, wn + 0);
    v16bf p1 = load_b_frag(B1sw, wn + 16);
    v16bf q0 = load_b_frag(B2sw, wn + 0);
    v16bf q1 = load_b_frag(B2sw, wn + 16);
    acc1[0] = wmma_bf16(a0, p0, acc1[0]);
    acc1[1] = wmma_bf16(a0, p1, acc1[1]);
    acc1[2] = wmma_bf16(a1, p0, acc1[2]);
    acc1[3] = wmma_bf16(a1, p1, acc1[3]);
    acc2[0] = wmma_bf16(a0, q0, acc2[0]);
    acc2[1] = wmma_bf16(a0, q1, acc2[1]);
    acc2[2] = wmma_bf16(a1, q0, acc2[2]);
    acc2[3] = wmma_bf16(a1, q1, acc2[3]);
    __syncthreads();
  }
  (void)M;
  int lane = threadIdx.x & 31, half = lane >> 4, coll = lane & 15;
#pragma unroll
  for (int t = 0; t < 4; ++t){
    int mbase = m0 + wm + (t >> 1)*16;
    int col   = n0 + wn + (t & 1)*16 + coll;
    FragF f1; f1.v = acc1[t];
    FragF f2; f2.v = acc2[t];
    float bb1 = b1[col], bb2 = b2[col];
    unsigned short tmp[8];
#pragma unroll
    for (int r = 0; r < 8; ++r)
      tmp[r] = f2bf((f1.f[r] + bb1) * sigm(f2.f[r] + bb2));
    unsigned base = (unsigned)col*uLL + (unsigned)(mbase + half*8);   // 8 consecutive rows
    *reinterpret_cast<uint4*>(&Gt[base]) = *reinterpret_cast<const uint4*>(tmp);
  }
}

// ---------------- triangle einsum on CHANNEL-MAJOR bf16 inputs, f32 row-major out ----------
// Lt/Rt: X_t[c*cLL + x*cL + y] == original X[x, y, c].
// OUTGOING: out[i,j,c] = sum_k Lt[c][i*cL+k] * Rt[c][j*cL+k]   (both contiguous in k)
// INCOMING: out[i,j,c] = sum_k Lt[c][k*cL+i] * Rt[c][k*cL+j]   (contiguous in i/j)
template<bool OUTGOING>
__global__ __launch_bounds__(128) void tri_einsum_kernel(
    const unsigned short* __restrict__ Lt, const unsigned short* __restrict__ Rt,
    float* __restrict__ Out)
{
  __shared__ alignas(16) unsigned short As[64*32];   // [i][k]
  __shared__ alignas(16) unsigned short Bsw[64*32];  // [j][k]
  int c  = blockIdx.x >> 6;
  int t  = blockIdx.x & 63;
  int i0 = (t >> 3)*64, j0 = (t & 7)*64;
  unsigned cbase = (unsigned)c * uLL;
  int wave = threadIdx.x >> 5;
  int wm = (wave >> 1)*32, wn = (wave & 1)*32;
  v8f acc[4] = {};
  for (int k0 = 0; k0 < cL; k0 += 32){
    if (OUTGOING){
#pragma unroll
      for (int it = 0; it < 4; ++it){
        unsigned idx = (unsigned)threadIdx.x + 128u*(unsigned)it;
        unsigned r = idx >> 3, kq = (idx & 7) << 2;
        const unsigned short* srcA = &Lt[cbase + (unsigned)(i0 + (int)r)*(unsigned)cL + (unsigned)k0 + kq];
        const unsigned short* srcB = &Rt[cbase + (unsigned)(j0 + (int)r)*(unsigned)cL + (unsigned)k0 + kq];
#if ASYNC_LDS
        cp8_async(&As[r*32 + kq],  srcA);
        cp8_async(&Bsw[r*32 + kq], srcB);
#else
        *reinterpret_cast<uint2*>(&As[r*32 + kq])  = *reinterpret_cast<const uint2*>(srcA);
        *reinterpret_cast<uint2*>(&Bsw[r*32 + kq]) = *reinterpret_cast<const uint2*>(srcB);
#endif
      }
#if ASYNC_LDS
      async_wait0();
#endif
    } else {
      // vectorized along rows (i/j contiguous): 8B loads, scatter 4x ds_store_b16
#pragma unroll
      for (int it = 0; it < 4; ++it){
        unsigned idx = (unsigned)threadIdx.x + 128u*(unsigned)it;
        unsigned kk = idx >> 4, rq = (idx & 15) << 2;
        uint2 w = *reinterpret_cast<const uint2*>(&Lt[cbase + (unsigned)(k0 + (int)kk)*(unsigned)cL + (unsigned)(i0 + (int)rq)]);
        const unsigned short* ws = reinterpret_cast<const unsigned short*>(&w);
        As[(rq+0)*32 + kk] = ws[0]; As[(rq+1)*32 + kk] = ws[1];
        As[(rq+2)*32 + kk] = ws[2]; As[(rq+3)*32 + kk] = ws[3];
      }
#pragma unroll
      for (int it = 0; it < 4; ++it){
        unsigned idx = (unsigned)threadIdx.x + 128u*(unsigned)it;
        unsigned kk = idx >> 4, nq = (idx & 15) << 2;
        uint2 w = *reinterpret_cast<const uint2*>(&Rt[cbase + (unsigned)(k0 + (int)kk)*(unsigned)cL + (unsigned)(j0 + (int)nq)]);
        const unsigned short* ws = reinterpret_cast<const unsigned short*>(&w);
        Bsw[(nq+0)*32 + kk] = ws[0]; Bsw[(nq+1)*32 + kk] = ws[1];
        Bsw[(nq+2)*32 + kk] = ws[2]; Bsw[(nq+3)*32 + kk] = ws[3];
      }
    }
    __syncthreads();
    v16bf a0 = load_a_frag(As, wm + 0);
    v16bf a1 = load_a_frag(As, wm + 16);
    v16bf b0 = load_b_frag(Bsw, wn + 0);
    v16bf b1 = load_b_frag(Bsw, wn + 16);
    acc[0] = wmma_bf16(a0, b0, acc[0]);
    acc[1] = wmma_bf16(a0, b1, acc[1]);
    acc[2] = wmma_bf16(a1, b0, acc[2]);
    acc[3] = wmma_bf16(a1, b1, acc[3]);
    __syncthreads();
  }
  int lane = threadIdx.x & 31, half = lane >> 4, coll = lane & 15;
#pragma unroll
  for (int t4 = 0; t4 < 4; ++t4){
    int mbase = wm + (t4 >> 1)*16;
    int col   = wn + (t4 & 1)*16 + coll;
    FragF fr; fr.v = acc[t4];
#pragma unroll
    for (int r = 0; r < 8; ++r){
      int row = mbase + half*8 + r;
      unsigned offc = ((unsigned)(i0 + row)*(unsigned)cL + (unsigned)(j0 + col))*(unsigned)cPD + (unsigned)c;
      Out[offc] = fr.f[r];
    }
  }
}

// ---------------- LayerNorm (row-wise), f32 in; f32 or bf16 out ----------------
template<bool OUTBF>
__global__ __launch_bounds__(128) void ln_kernel(
    const float* __restrict__ x, const float* __restrict__ g, const float* __restrict__ b,
    float* yf, unsigned short* yb, int D)
{
  unsigned base = (unsigned)blockIdx.x * (unsigned)D;
  const float* xr = x + base;
  int tid = threadIdx.x;
  float s = 0.f, s2 = 0.f;
  for (int d = tid; d < D; d += 128){ float v = xr[d]; s += v; s2 += v*v; }
  __shared__ float r1[128], r2[128];
  r1[tid] = s; r2[tid] = s2; __syncthreads();
  for (int o = 64; o > 0; o >>= 1){
    if (tid < o){ r1[tid] += r1[tid+o]; r2[tid] += r2[tid+o]; }
    __syncthreads();
  }
  float mean = r1[0] / (float)D;
  float var  = r2[0] / (float)D - mean*mean;
  float inv  = rsqrtf(var + 1e-5f);
  for (int d = tid; d < D; d += 128){
    float v = (xr[d] - mean)*inv*g[d] + b[d];
    if (OUTBF) yb[base + d] = f2bf(v);
    else       yf[base + d] = v;
  }
}

// ---------------- softmax over last dim (512) in place ----------------
__global__ __launch_bounds__(128) void softmax_kernel(float* __restrict__ a){
  float* p = a + (unsigned)blockIdx.x * (unsigned)cL;
  int tid = threadIdx.x;
  __shared__ float red[128];
  float m = -1e30f;
  for (int j = tid; j < cL; j += 128) m = fmaxf(m, p[j]);
  red[tid] = m; __syncthreads();
  for (int o = 64; o > 0; o >>= 1){ if (tid < o) red[tid] = fmaxf(red[tid], red[tid+o]); __syncthreads(); }
  m = red[0]; __syncthreads();
  float s = 0.f;
  for (int j = tid; j < cL; j += 128){ float e = __expf(p[j] - m); p[j] = e; s += e; }
  red[tid] = s; __syncthreads();
  for (int o = 64; o > 0; o >>= 1){ if (tid < o) red[tid] += red[tid+o]; __syncthreads(); }
  float invs = 1.f / red[0];
  for (int j = tid; j < cL; j += 128) p[j] *= invs;
}

// ---------------- elementwise helpers ----------------
__global__ void f2bf_kernel(const float* __restrict__ in, unsigned short* __restrict__ out, size_t n){
  size_t i = (size_t)blockIdx.x*blockDim.x + threadIdx.x;
  size_t st = (size_t)gridDim.x*blockDim.x;
  for (; i < n; i += st) out[i] = f2bf(in[i]);
}
__global__ void copy_kernel(const float* __restrict__ in, float* __restrict__ out, size_t n){
  size_t i = (size_t)blockIdx.x*blockDim.x + threadIdx.x;
  size_t st = (size_t)gridDim.x*blockDim.x;
  for (; i < n; i += st) out[i] = in[i];
}
__global__ void add_time_kernel(const float* __restrict__ node, const float* __restrict__ t,
                                float* __restrict__ out, size_t n){
  size_t i = (size_t)blockIdx.x*blockDim.x + threadIdx.x;
  size_t st = (size_t)gridDim.x*blockDim.x;
  for (; i < n; i += st) out[i] = node[i] + t[i % cND];
}
__global__ void pair_update_kernel(float* __restrict__ pair, const float* __restrict__ gate,
                                   const float* __restrict__ o, size_t n){
  size_t i = (size_t)blockIdx.x*blockDim.x + threadIdx.x;
  size_t st = (size_t)gridDim.x*blockDim.x;
  for (; i < n; i += st) pair[i] += gate[i]*o[i];
}

// ---------------- time MLP (384 -> silu -> 384), one block of 384 threads ----------------
__global__ void time_mlp_kernel(const float* __restrict__ te_in,
                                const float* __restrict__ w1, const float* __restrict__ b1,
                                const float* __restrict__ w2, const float* __restrict__ b2,
                                float* __restrict__ t_out){
  __shared__ float te[cND];
  __shared__ float t1[cND];
  int tid = threadIdx.x;
  te[tid] = te_in[tid];
  __syncthreads();
  float acc = b1[tid];
  for (int i = 0; i < cND; ++i) acc += te[i]*w1[i*cND + tid];
  t1[tid] = acc * sigm(acc);     // silu
  __syncthreads();
  float acc2 = b2[tid];
  for (int i = 0; i < cND; ++i) acc2 += t1[i]*w2[i*cND + tid];
  t_out[tid] = acc2;
}

// ---------------- apply frames: out = R*p + t, one thread per (l, point) ----------------
__global__ void frames_kernel(const float* __restrict__ pts, const float* __restrict__ rots,
                              const float* __restrict__ trans, float* __restrict__ out,
                              int ppl, int total){
  int idx = blockIdx.x*blockDim.x + threadIdx.x;
  if (idx >= total) return;
  int l = idx / ppl;
  const float* R = rots + l*9;
  const float* t = trans + l*3;
  const float* p = pts + (size_t)idx*3;
  float* o = out + (size_t)idx*3;
  float p0 = p[0], p1 = p[1], p2 = p[2];
  o[0] = R[0]*p0 + R[1]*p1 + R[2]*p2 + t[0];
  o[1] = R[3]*p0 + R[4]*p1 + R[5]*p2 + t[1];
  o[2] = R[6]*p0 + R[7]*p1 + R[8]*p2 + t[2];
}

// ---------------- attention logits ----------------
__global__ void logits_kernel(const float* __restrict__ q, const float* __restrict__ k,
                              const float* __restrict__ qg, const float* __restrict__ kg,
                              const float* __restrict__ pair, const float* __restrict__ wpb,
                              const float* __restrict__ hw, float* __restrict__ logits){
  int idx = blockIdx.x*blockDim.x + threadIdx.x;
  if (idx >= cH*cL*cL) return;
  int h = idx / (cL*cL);
  int rem = idx % (cL*cL);
  int i = rem / cL, j = rem % cL;
  const float* qi = q + (unsigned)i*cND + h*cHD;
  const float* kj = k + (unsigned)j*cND + h*cHD;
  float acc = 0.f;
  for (int d = 0; d < cHD; ++d) acc += qi[d]*kj[d];
  acc *= 0.14433756729740643f;   // 1/sqrt(48)
  const float* qgi = qg + (unsigned)i*(cH*cNQK*3) + h*cNQK*3;
  const float* kgj = kg + (unsigned)j*(cH*cNQK*3) + h*cNQK*3;
  float dist = 0.f;
  for (int t = 0; t < cNQK*3; ++t){ float d = qgi[t] - kgj[t]; dist += d*d; }
  float wc = logf(1.f + __expf(hw[h]));  // softplus
  const float* pij = pair + ((unsigned)i*cL + (unsigned)j)*(unsigned)cPD;
  float pb = 0.f;
  for (int c = 0; c < cPD; ++c) pb += pij[c]*wpb[c*cH + h];
  logits[idx] = acc - 0.5f*wc*dist + pb;
}

// ---------------- attention weighted sums ----------------
__global__ void attn_scalar_kernel(const float* __restrict__ a, const float* __restrict__ v,
                                   float* __restrict__ outcat){
  int idx = blockIdx.x*blockDim.x + threadIdx.x;
  if (idx >= cL*cND) return;
  int i = idx / cND, r = idx % cND, h = r / cHD;
  const float* ar = a + (unsigned)h*uLL + (unsigned)i*cL;
  float acc = 0.f;
  for (int j = 0; j < cL; ++j) acc += ar[j]*v[(unsigned)j*cND + r];
  outcat[(unsigned)i*cOC + r] = acc;
}
__global__ void attn_ptsg_kernel(const float* __restrict__ a, const float* __restrict__ vg,
                                 float* __restrict__ og){
  int idx = blockIdx.x*blockDim.x + threadIdx.x;
  if (idx >= cL*cH*cNV*3) return;
  int i = idx / (cH*cNV*3), r = idx % (cH*cNV*3), h = r / (cNV*3);
  const float* ar = a + (unsigned)h*uLL + (unsigned)i*cL;
  float acc = 0.f;
  for (int j = 0; j < cL; ++j) acc += ar[j]*vg[(unsigned)j*(cH*cNV*3) + r];
  og[idx] = acc;
}
__global__ void attn_pts_local_kernel(const float* __restrict__ og, const float* __restrict__ rots,
                                      const float* __restrict__ trans, float* __restrict__ outcat){
  int idx = blockIdx.x*blockDim.x + threadIdx.x;
  if (idx >= cL*cH*cNV) return;
  int l = idx / (cH*cNV), qp = idx % (cH*cNV);
  const float* R = rots + l*9;
  const float* t = trans + l*3;
  const float* p = og + ((unsigned)l*(cH*cNV) + qp)*3;
  float d0 = p[0]-t[0], d1 = p[1]-t[1], d2 = p[2]-t[2];
  float* o = outcat + (unsigned)l*cOC + cND + qp*3;
  o[0] = R[0]*d0 + R[3]*d1 + R[6]*d2;   // R^T
  o[1] = R[1]*d0 + R[4]*d1 + R[7]*d2;
  o[2] = R[2]*d0 + R[5]*d1 + R[8]*d2;
}
__global__ void attn_pair_kernel(const float* __restrict__ a, const float* __restrict__ pair,
                                 float* __restrict__ outcat){
  int idx = blockIdx.x*blockDim.x + threadIdx.x;
  if (idx >= cL*cH*cPD) return;
  int i = idx / (cH*cPD), r = idx % (cH*cPD), h = r / cPD, c = r % cPD;
  const float* ar = a + (unsigned)h*uLL + (unsigned)i*cL;
  const float* pr = pair + ((unsigned)i*cL)*(unsigned)cPD + (unsigned)c;
  float acc = 0.f;
  for (int j = 0; j < cL; ++j) acc += ar[j]*pr[(unsigned)j*cPD];
  outcat[(unsigned)i*cOC + cND + cH*cNV*3 + r] = acc;
}

// ---------------- frame update (LN + linear(6) + axis-angle + compose) ----------------
__global__ __launch_bounds__(64) void frame_update_kernel(
    const float* __restrict__ node, const float* __restrict__ g, const float* __restrict__ b,
    const float* __restrict__ w, const float* __restrict__ b6,
    const float* __restrict__ rots, const float* __restrict__ trans,
    float* __restrict__ orots, float* __restrict__ otrans)
{
  int l = blockIdx.x, tid = threadIdx.x;
  const float* x = node + (size_t)l*cND;
  __shared__ float red[64];
  float s = 0.f, s2 = 0.f;
  for (int d = tid; d < cND; d += 64){ float v = x[d]; s += v; s2 += v*v; }
  red[tid] = s; __syncthreads();
  for (int o = 32; o > 0; o >>= 1){ if (tid < o) red[tid] += red[tid+o]; __syncthreads(); }
  float mean = red[0] / (float)cND; __syncthreads();
  red[tid] = s2; __syncthreads();
  for (int o = 32; o > 0; o >>= 1){ if (tid < o) red[tid] += red[tid+o]; __syncthreads(); }
  float var = red[0] / (float)cND - mean*mean; __syncthreads();
  float inv = rsqrtf(var + 1e-5f);
  float p[6] = {0.f,0.f,0.f,0.f,0.f,0.f};
  for (int d = tid; d < cND; d += 64){
    float xn = (x[d] - mean)*inv*g[d] + b[d];
    for (int j = 0; j < 6; ++j) p[j] += xn*w[d*6 + j];
  }
  __shared__ float red6[6][64];
  for (int j = 0; j < 6; ++j) red6[j][tid] = p[j];
  __syncthreads();
  for (int o = 32; o > 0; o >>= 1){
    if (tid < o) for (int j = 0; j < 6; ++j) red6[j][tid] += red6[j][tid+o];
    __syncthreads();
  }
  if (tid == 0){
    float u[6];
    for (int j = 0; j < 6; ++j) u[j] = red6[j][0] + b6[j];
    float ax = u[0]*0.1f, ay = u[1]*0.1f, az = u[2]*0.1f;
    float n  = sqrtf(ax*ax + ay*ay + az*az);
    float ux = ax/(n+1e-8f), uy = ay/(n+1e-8f), uz = az/(n+1e-8f);
    float sn = sinf(n), cc = 1.f - cosf(n);
    float R[9];
    R[0] = 1.f + cc*(-(uy*uy+uz*uz)); R[1] = -sn*uz + cc*(ux*uy);       R[2] =  sn*uy + cc*(ux*uz);
    R[3] =  sn*uz + cc*(ux*uy);       R[4] = 1.f + cc*(-(ux*ux+uz*uz)); R[5] = -sn*ux + cc*(uy*uz);
    R[6] = -sn*uy + cc*(ux*uz);       R[7] =  sn*ux + cc*(uy*uz);       R[8] = 1.f + cc*(-(ux*ux+uy*uy));
    const float* Ri = rots + l*9;
    float* Ro = orots + l*9;
    for (int i = 0; i < 3; ++i)
      for (int j = 0; j < 3; ++j)
        Ro[i*3+j] = Ri[i*3+0]*R[0*3+j] + Ri[i*3+1]*R[1*3+j] + Ri[i*3+2]*R[2*3+j];
    float* to = otrans + l*3;
    for (int i = 0; i < 3; ++i)
      to[i] = Ri[i*3+0]*u[3] + Ri[i*3+1]*u[4] + Ri[i*3+2]*u[5] + trans[l*3+i];
  }
}

// =========================================================================================
extern "C" void kernel_launch(void* const* d_in, const int* in_sizes, int n_in,
                              void* d_out, int out_size, void* d_ws, size_t ws_size,
                              hipStream_t stream)
{
  (void)in_sizes; (void)n_in; (void)out_size; (void)ws_size;
  auto F = [&](int i)->const float*{ return (const float*)d_in[i]; };

  const float* node_in = F(IN_NODE);
  const float* pair_in = F(IN_PAIR);
  const float* rots_in = F(IN_ROTS);
  const float* trans_in = F(IN_TRANS);

  float* out_node  = (float*)d_out;
  float* out_pair  = out_node + (size_t)cL*cND;
  float* out_rots  = out_pair + cLL*cPD;
  float* out_trans = out_rots + (size_t)cL*9;

  // ---- workspace bump allocator ----
  char* wsb = (char*)d_ws; size_t off = 0;
  auto alloc = [&](size_t bytes)->void*{
    void* p = wsb + off; off = (off + bytes + 255) & ~(size_t)255; return p;
  };
  float* t_out    = (float*)alloc(cND*4);
  float* node_tmp = (float*)alloc((size_t)cL*cND*4);
  float* s        = (float*)alloc((size_t)cL*cND*4);
  unsigned short* s2bf = (unsigned short*)alloc((size_t)cL*cND*2);
  float* q  = (float*)alloc((size_t)cL*cND*4);
  float* k  = (float*)alloc((size_t)cL*cND*4);
  float* v  = (float*)alloc((size_t)cL*cND*4);
  float* qpts = (float*)alloc((size_t)cL*cH*cNQK*3*4);
  float* kpts = (float*)alloc((size_t)cL*cH*cNQK*3*4);
  float* vpts = (float*)alloc((size_t)cL*cH*cNV*3*4);
  float* qg = (float*)alloc((size_t)cL*cH*cNQK*3*4);
  float* kg = (float*)alloc((size_t)cL*cH*cNQK*3*4);
  float* vg = (float*)alloc((size_t)cL*cH*cNV*3*4);
  float* og = (float*)alloc((size_t)cL*cH*cNV*3*4);
  float* logits = (float*)alloc((size_t)cH*cLL*4);
  float* outcat = (float*)alloc((size_t)cL*cOC*4);
  unsigned short* outcat_bf = (unsigned short*)alloc((size_t)cL*cOC*2);
  unsigned short* ntbf  = (unsigned short*)alloc((size_t)cL*cND*2);
  unsigned short* hid_n = (unsigned short*)alloc((size_t)cL*cND*4*2);  // 512x1536 bf16
  // bf16 weights
  unsigned short* wq_bf  = (unsigned short*)alloc((size_t)cND*cND*2);
  unsigned short* wk_bf  = (unsigned short*)alloc((size_t)cND*cND*2);
  unsigned short* wv_bf  = (unsigned short*)alloc((size_t)cND*cND*2);
  unsigned short* wqp_bf = (unsigned short*)alloc((size_t)cND*cH*cNQK*3*2);
  unsigned short* wkp_bf = (unsigned short*)alloc((size_t)cND*cH*cNQK*3*2);
  unsigned short* wvp_bf = (unsigned short*)alloc((size_t)cND*cH*cNV*3*2);
  unsigned short* woipa_bf = (unsigned short*)alloc((size_t)cOC*cND*2);
  unsigned short* tw1_bf = (unsigned short*)alloc((size_t)cND*cND*4*2);
  unsigned short* tw2_bf = (unsigned short*)alloc((size_t)cND*4*cND*2);
  unsigned short* pw1_bf = (unsigned short*)alloc((size_t)cPD*cPD*4*2);
  unsigned short* pw2_bf = (unsigned short*)alloc((size_t)cPD*4*cPD*2);
  unsigned short* tri_w_bf[12];
  for (int i = 0; i < 12; ++i) tri_w_bf[i] = (unsigned short*)alloc((size_t)cPD*cPD*2);
  // pair-sized buffers
  unsigned short* z_bf  = (unsigned short*)alloc(cLL*cPD*2);
  unsigned short* lg_bf = (unsigned short*)alloc(cLL*cPD*2);   // channel-major gated L / later row-major reuse
  unsigned short* rg_bf = (unsigned short*)alloc(cLL*cPD*2);   // channel-major gated R
  float* big0 = (float*)alloc(cLL*cPD*4);
  float* big1 = (float*)alloc(cLL*cPD*4);
  unsigned short* hid_p = (unsigned short*)big0;  // 262144x512 bf16 spans big0+big1

  auto ewb = [&](size_t n)->int{ size_t bl = (n + 255)/256; return (int)(bl > 32768 ? 32768 : bl); };
  auto conv = [&](const float* src, unsigned short* dst, size_t n){
    f2bf_kernel<<<ewb(n), 256, 0, stream>>>(src, dst, n);
  };
  auto gblocks = [](int M, int N)->int{ return ((M+63)/64)*((N+63)/64); };

  // ---- convert weights to bf16 ----
  conv(F(IPA_WQ), wq_bf, (size_t)cND*cND);
  conv(F(IPA_WK), wk_bf, (size_t)cND*cND);
  conv(F(IPA_WV), wv_bf, (size_t)cND*cND);
  conv(F(IPA_WQ_PTS), wqp_bf, (size_t)cND*cH*cNQK*3);
  conv(F(IPA_WK_PTS), wkp_bf, (size_t)cND*cH*cNQK*3);
  conv(F(IPA_WV_PTS), wvp_bf, (size_t)cND*cH*cNV*3);
  conv(F(IPA_WO), woipa_bf, (size_t)cOC*cND);
  conv(F(NT_W1), tw1_bf, (size_t)cND*cND*4);
  conv(F(NT_W2), tw2_bf, (size_t)cND*4*cND);
  conv(F(PT_W1), pw1_bf, (size_t)cPD*cPD*4);
  conv(F(PT_W2), pw2_bf, (size_t)cPD*4*cPD);
  const int to_w[6] = {TO_WL, TO_WLG, TO_WR, TO_WRG, TO_WO, TO_WOG};
  const int ti_w[6] = {TI_WL, TI_WLG, TI_WR, TI_WRG, TI_WO, TI_WOG};
  for (int i = 0; i < 6; ++i){
    conv(F(to_w[i]), tri_w_bf[i],     (size_t)cPD*cPD);
    conv(F(ti_w[i]), tri_w_bf[6+i],   (size_t)cPD*cPD);
  }

  // ---- time embed MLP + add ----
  time_mlp_kernel<<<1, cND, 0, stream>>>(F(IN_TEMBED), F(TIME_W1), F(TIME_B1), F(TIME_W2), F(TIME_B2), t_out);
  add_time_kernel<<<ewb((size_t)cL*cND), 256, 0, stream>>>(node_in, t_out, node_tmp, (size_t)cL*cND);

  // ---- IPA ----
  ln_kernel<false><<<cL, 128, 0, stream>>>(node_tmp, F(P_IPA_NORM_G), F(P_IPA_NORM_B), s, nullptr, cND);
  ln_kernel<true ><<<cL, 128, 0, stream>>>(s, F(IPA_LN_G), F(IPA_LN_B), nullptr, s2bf, cND);

  gemm_kernel<false,0,0><<<gblocks(cL,cND), 128, 0, stream>>>(s2bf, wq_bf, nullptr, q, nullptr, cL, cND, cND);
  gemm_kernel<false,0,0><<<gblocks(cL,cND), 128, 0, stream>>>(s2bf, wk_bf, nullptr, k, nullptr, cL, cND, cND);
  gemm_kernel<false,0,0><<<gblocks(cL,cND), 128, 0, stream>>>(s2bf, wv_bf, nullptr, v, nullptr, cL, cND, cND);
  gemm_kernel<true ,0,0><<<gblocks(cL,cH*cNQK*3), 128, 0, stream>>>(s2bf, wqp_bf, F(IPA_BQ_PTS), qpts, nullptr, cL, cH*cNQK*3, cND);
  gemm_kernel<true ,0,0><<<gblocks(cL,cH*cNQK*3), 128, 0, stream>>>(s2bf, wkp_bf, F(IPA_BK_PTS), kpts, nullptr, cL, cH*cNQK*3, cND);
  gemm_kernel<true ,0,0><<<gblocks(cL,cH*cNV*3), 128, 0, stream>>>(s2bf, wvp_bf, F(IPA_BV_PTS), vpts, nullptr, cL, cH*cNV*3, cND);

  frames_kernel<<<(cL*cH*cNQK + 255)/256, 256, 0, stream>>>(qpts, rots_in, trans_in, qg, cH*cNQK, cL*cH*cNQK);
  frames_kernel<<<(cL*cH*cNQK + 255)/256, 256, 0, stream>>>(kpts, rots_in, trans_in, kg, cH*cNQK, cL*cH*cNQK);
  frames_kernel<<<(cL*cH*cNV  + 255)/256, 256, 0, stream>>>(vpts, rots_in, trans_in, vg, cH*cNV,  cL*cH*cNV);

  logits_kernel<<<(int)((cH*cLL + 255)/256), 256, 0, stream>>>(q, k, qg, kg, pair_in, F(IPA_W_PAIR_BIAS), F(IPA_HEAD_W), logits);
  softmax_kernel<<<cH*cL, 128, 0, stream>>>(logits);

  attn_scalar_kernel<<<(cL*cND + 255)/256, 256, 0, stream>>>(logits, v, outcat);
  attn_ptsg_kernel<<<(cL*cH*cNV*3 + 255)/256, 256, 0, stream>>>(logits, vg, og);
  attn_pts_local_kernel<<<(cL*cH*cNV + 255)/256, 256, 0, stream>>>(og, rots_in, trans_in, outcat);
  attn_pair_kernel<<<(cL*cH*cPD + 255)/256, 256, 0, stream>>>(logits, pair_in, outcat);

  conv(outcat, outcat_bf, (size_t)cL*cOC);
  gemm_kernel<true,0,1><<<gblocks(cL,cND), 128, 0, stream>>>(outcat_bf, woipa_bf, F(IPA_BO), node_tmp, nullptr, cL, cND, cOC);

  // ---- node transition ----
  ln_kernel<true><<<cL, 128, 0, stream>>>(node_tmp, F(NT_G), F(NT_B), nullptr, ntbf, cND);
  gemm_kernel<true,1,2><<<gblocks(cL,cND*4), 128, 0, stream>>>(ntbf, tw1_bf, F(NT_B1), nullptr, hid_n, cL, cND*4, cND);
  gemm_kernel<true,0,1><<<gblocks(cL,cND), 128, 0, stream>>>(hid_n, tw2_bf, F(NT_B2), node_tmp, nullptr, cL, cND, cND*4);

  copy_kernel<<<ewb((size_t)cL*cND), 256, 0, stream>>>(node_tmp, out_node, (size_t)cL*cND);
  frame_update_kernel<<<cL, 64, 0, stream>>>(node_tmp, F(P_FRAME_NORM_G), F(P_FRAME_NORM_B),
      F(P_FRAME_UPD_W), F(P_FRAME_UPD_B), rots_in, trans_in, out_rots, out_trans);

  // ---- pair track ----
  copy_kernel<<<32768, 256, 0, stream>>>(pair_in, out_pair, cLL*cPD);

  auto tri = [&](bool outgoing, int LNG, int LNB, int BL, int BLG, int BR, int BRG,
                 int FNG, int FNB, int BO, int BOG, unsigned short** W){
    const int M = (int)cLL;
    ln_kernel<true><<<M, 128, 0, stream>>>(out_pair, F(LNG), F(LNB), nullptr, z_bf, cPD);
    gemm_gated_kernel<<<(M/64)*2, 128, 0, stream>>>(z_bf, W[0], W[1], F(BL), F(BLG), lg_bf, M);
    gemm_gated_kernel<<<(M/64)*2, 128, 0, stream>>>(z_bf, W[2], W[3], F(BR), F(BRG), rg_bf, M);
    if (outgoing) tri_einsum_kernel<true ><<<cPD*64, 128, 0, stream>>>(lg_bf, rg_bf, big0);
    else          tri_einsum_kernel<false><<<cPD*64, 128, 0, stream>>>(lg_bf, rg_bf, big0);
    ln_kernel<true><<<M, 128, 0, stream>>>(big0, F(FNG), F(FNB), nullptr, z_bf, cPD);
    gemm_kernel<true,0,0><<<gblocks(M,cPD), 128, 0, stream>>>(z_bf, W[4], F(BO), big1, nullptr, M, cPD, cPD);
    conv(out_pair, lg_bf, cLL*cPD);
    gemm_kernel<true,2,0><<<gblocks(M,cPD), 128, 0, stream>>>(lg_bf, W[5], F(BOG), big0, nullptr, M, cPD, cPD);
    pair_update_kernel<<<32768, 256, 0, stream>>>(out_pair, big0, big1, cLL*cPD);
  };
  unsigned short* Wto[6] = {tri_w_bf[0], tri_w_bf[1], tri_w_bf[2], tri_w_bf[3], tri_w_bf[4], tri_w_bf[5]};
  unsigned short* Wti[6] = {tri_w_bf[6], tri_w_bf[7], tri_w_bf[8], tri_w_bf[9], tri_w_bf[10], tri_w_bf[11]};
  tri(true,  TO_LN_G, TO_LN_B, TO_BL, TO_BLG, TO_BR, TO_BRG, TO_FN_G, TO_FN_B, TO_BO, TO_BOG, Wto);
  tri(false, TI_LN_G, TI_LN_B, TI_BL, TI_BLG, TI_BR, TI_BRG, TI_FN_G, TI_FN_B, TI_BO, TI_BOG, Wti);

  // ---- pair transition ----
  {
    const int M = (int)cLL;
    ln_kernel<true><<<M, 128, 0, stream>>>(out_pair, F(PT_G), F(PT_B), nullptr, z_bf, cPD);
    gemm_kernel<true,1,2><<<gblocks(M,cPD*4), 128, 0, stream>>>(z_bf, pw1_bf, F(PT_B1), nullptr, hid_p, M, cPD*4, cPD);
    gemm_kernel<true,0,1><<<gblocks(M,cPD), 128, 0, stream>>>(hid_p, pw2_bf, F(PT_B2), out_pair, nullptr, M, cPD, cPD*4);
  }
}
